// QuantDenseModelLarge_44487271252574
// MI455X (gfx1250) — 16-bit and 8-bit matrix operand order repaired and checked by reading, compiled, not run on hardware
//
#include <hip/hip_runtime.h>
#include <stdint.h>

typedef __attribute__((ext_vector_type(16))) _Float16 v16h;
typedef __attribute__((ext_vector_type(8)))  float    v8f;
typedef __attribute__((ext_vector_type(8)))  int      v8i;
typedef __attribute__((ext_vector_type(4)))  float    v4f;
typedef __attribute__((ext_vector_type(4)))  int      v4i;

#define WS_W1P    0
#define WS_W2P    32768
#define WS_W3P    36864
#define WS_WOP    40960
#define WS_SCALES 41984

#define W1_LDS_STRIDE 264
#define ST_STRIDE     72

__global__ __launch_bounds__(256) void qdense_prep(
    const float* __restrict__ W1, const float* __restrict__ W2,
    const float* __restrict__ W3, const float* __restrict__ Wo,
    uint8_t* __restrict__ ws)
{
    __shared__ float red[256];
    const int t = threadIdx.x;
    const int b = blockIdx.x;

    const float* W; int rows, cols;
    if      (b == 0) { W = W1; rows = 58; cols = 256; }
    else if (b == 1) { W = W2; rows = 58; cols = 58;  }
    else if (b == 2) { W = W3; rows = 58; cols = 58;  }
    else             { W = Wo; rows = 12; cols = 58;  }

    const int n = rows * cols;
    float m = 0.0f;
    for (int i = t; i < n; i += 256) m = fmaxf(m, fabsf(W[i]));
    red[t] = m;
    __syncthreads();
    #pragma unroll
    for (int s = 128; s > 0; s >>= 1) {
        if (t < s) red[t] = fmaxf(red[t], red[t + s]);
        __syncthreads();
    }
    const float scale = red[0] * (1.0f / 7.0f);
    const float inv   = 7.0f / red[0];
    if (t == 0) ((float*)(ws + WS_SCALES))[b] = scale;

    if (b == 0) {
        _Float16* out = (_Float16*)(ws + WS_W1P);
        for (int i = t; i < 64 * 256; i += 256) {
            const int r = i >> 8;
            float v = 0.0f;
            if (r < 58) {
                float q = rintf(W[i] * inv);
                q = fminf(fmaxf(q, -7.0f), 7.0f);
                v = q * scale;
            }
            out[i] = (_Float16)v;
        }
    } else {
        int8_t* out = (int8_t*)(ws + (b == 1 ? WS_W2P : (b == 2 ? WS_W3P : WS_WOP)));
        const int rows_pad = (b == 3) ? 16 : 64;
        for (int i = t; i < rows_pad * 64; i += 256) {
            const int r = i >> 6, c = i & 63;
            int q = 0;
            if (r < rows && c < cols) {
                float qf = rintf(W[r * cols + c] * inv);
                qf = fminf(fmaxf(qf, -7.0f), 7.0f);
                q = (int)qf;
            }
            out[i] = (int8_t)q;
        }
    }
}

__device__ __forceinline__ v8i load_a_u8(const uint8_t* st, int ln, int hi) {
    v8i A;
    #pragma unroll
    for (int j = 0; j < 4; j++) {
        const uint64_t d = *(const uint64_t*)(st + ln * ST_STRIDE + (j >> 1) * 32 + (j & 1) * 8 + hi * 16);
        A[2 * j]     = (int)(uint32_t)(d & 0xffffffffull);
        A[2 * j + 1] = (int)(uint32_t)(d >> 32);
    }
    return A;
}

__global__ __launch_bounds__(256) void qdense_mlp(
    const float* __restrict__ x,
    const float* __restrict__ g1, const float* __restrict__ b1, const float* __restrict__ s1p,
    const float* __restrict__ g2, const float* __restrict__ b2, const float* __restrict__ s2p,
    const float* __restrict__ g3, const float* __restrict__ b3, const float* __restrict__ s3p,
    const float* __restrict__ bo,
    const uint8_t* __restrict__ ws,
    float* __restrict__ out)
{
    __shared__ _Float16 ldsW1[64 * W1_LDS_STRIDE];
    __shared__ uint8_t  stage[8][16 * ST_STRIDE];

    const int t    = threadIdx.x;
    const int wave = t >> 5;
    const int lane = t & 31;
    const int ln   = lane & 15;
    const int hi   = lane >> 4;

    {
        const uint32_t* src = (const uint32_t*)(ws + WS_W1P);
        for (int i = t; i < 64 * 128; i += 256) {
            const int r = i >> 7, c = i & 127;
            *(uint32_t*)&ldsW1[r * W1_LDS_STRIDE + c * 2] = src[i];
        }
    }

    const float* wsc = (const float*)(ws + WS_SCALES);
    const float s1 = s1p[0], s2 = s2p[0], s3 = s3p[0];
    const float is1 = 1.0f / s1, is2 = 1.0f / s2, is3 = 1.0f / s3;
    const float k2 = s1 * wsc[1];
    const float k3 = s2 * wsc[2];
    const float ko = s3 * wsc[3];

    v8i B2f[4], B3f[4], Bof;
    #pragma unroll
    for (int nt = 0; nt < 4; nt++) {
        const int row = nt * 16 + ln;
        const uint8_t* p2 = ws + WS_W2P + row * 64 + hi * 16;
        const uint8_t* p3 = ws + WS_W3P + row * 64 + hi * 16;
        v4i a2 = *(const v4i*)p2, c2 = *(const v4i*)(p2 + 32);
        v4i a3 = *(const v4i*)p3, c3 = *(const v4i*)(p3 + 32);
        B2f[nt] = __builtin_shufflevector(a2, c2, 0, 1, 2, 3, 4, 5, 6, 7);
        B3f[nt] = __builtin_shufflevector(a3, c3, 0, 1, 2, 3, 4, 5, 6, 7);
    }
    {
        const uint8_t* po = ws + WS_WOP + ln * 64 + hi * 16;
        v4i a = *(const v4i*)po, c = *(const v4i*)(po + 32);
        Bof = __builtin_shufflevector(a, c, 0, 1, 2, 3, 4, 5, 6, 7);
    }

    __syncthreads();

    const int rowBase = blockIdx.x * 128 + wave * 16;
    uint8_t* st = stage[wave];

    v8f C1[4];
    #pragma unroll
    for (int n = 0; n < 4; n++) C1[n] = {};

    const float* xr0 = x + (size_t)(rowBase + ln) * 256;
    #pragma unroll
    for (int k = 0; k < 8; k++) {
        const float* xr = xr0 + k * 32 + hi * 16;
        v4f p0 = __builtin_nontemporal_load((const v4f*)(xr));
        v4f p1 = __builtin_nontemporal_load((const v4f*)(xr + 4));
        v4f p2 = __builtin_nontemporal_load((const v4f*)(xr + 8));
        v4f p3 = __builtin_nontemporal_load((const v4f*)(xr + 12));
        v16h A;
        #pragma unroll
        for (int i = 0; i < 4; i++) {
            A[i]      = (_Float16)p0[i];
            A[4 + i]  = (_Float16)p1[i];
            A[8 + i]  = (_Float16)p2[i];
            A[12 + i] = (_Float16)p3[i];
        }
        #pragma unroll
        for (int n = 0; n < 4; n++) {
            const v16h Bf = *(const v16h*)&ldsW1[(n * 16 + ln) * W1_LDS_STRIDE + k * 32 + hi * 16];
            C1[n] = __builtin_amdgcn_wmma_f32_16x16x32_f16(false, A, false, Bf, (short)0,
                                                           C1[n], false, false);
        }
    }

    {
        float gv[4], bv[4];
        #pragma unroll
        for (int n = 0; n < 4; n++) {
            const int col = n * 16 + ln;
            gv[n] = (col < 58) ? g1[col] : 0.0f;
            bv[n] = (col < 58) ? b1[col] : 0.0f;
        }
        #pragma unroll
        for (int n = 0; n < 4; n++) {
            const int col = n * 16 + ln;
            #pragma unroll
            for (int r = 0; r < 8; r++) {
                const float y = C1[n][r] * gv[n] + bv[n];
                const float q = fminf(fmaxf(rintf(fmaxf(y, 0.0f) * is1), 0.0f), 255.0f);
                st[(r + 8 * hi) * ST_STRIDE + col] = (uint8_t)q;
            }
        }
    }

    const v8i zi = {};

    {
        const v8i A2 = load_a_u8(st, ln, hi);
        float gv[4], bv[4];
        #pragma unroll
        for (int n = 0; n < 4; n++) {
            const int col = n * 16 + ln;
            gv[n] = (col < 58) ? g2[col] * k2 : 0.0f;
            bv[n] = (col < 58) ? b2[col] : 0.0f;
        }
        #pragma unroll
        for (int n = 0; n < 4; n++) {
            const v8i C2 = __builtin_amdgcn_wmma_i32_16x16x64_iu8(false, A2, true, B2f[n],
                                                                  zi, false, false);
            const int col = n * 16 + ln;
            #pragma unroll
            for (int r = 0; r < 8; r++) {
                const float y = (float)C2[r] * gv[n] + bv[n];
                const float q = fminf(fmaxf(rintf(fmaxf(y, 0.0f) * is2), 0.0f), 255.0f);
                st[(r + 8 * hi) * ST_STRIDE + col] = (uint8_t)q;
            }
        }
    }

    {
        const v8i A3 = load_a_u8(st, ln, hi);
        float gv[4], bv[4];
        #pragma unroll
        for (int n = 0; n < 4; n++) {
            const int col = n * 16 + ln;
            gv[n] = (col < 58) ? g3[col] * k3 : 0.0f;
            bv[n] = (col < 58) ? b3[col] : 0.0f;
        }
        #pragma unroll
        for (int n = 0; n < 4; n++) {
            const v8i C3 = __builtin_amdgcn_wmma_i32_16x16x64_iu8(false, A3, true, B3f[n],
                                                                  zi, false, false);
            const int col = n * 16 + ln;
            #pragma unroll
            for (int r = 0; r < 8; r++) {
                const float y = (float)C3[r] * gv[n] + bv[n];
                const float q = fminf(fmaxf(rintf(fmaxf(y, 0.0f) * is3), 0.0f), 255.0f);
                st[(r + 8 * hi) * ST_STRIDE + col] = (uint8_t)q;
            }
        }
    }

    {
        const v8i A4 = load_a_u8(st, ln, hi);
        const v8i C4 = __builtin_amdgcn_wmma_i32_16x16x64_iu8(false, A4, true, Bof,
                                                              zi, false, false);
        if (ln < 12) {
            const float bv = bo[ln];
            #pragma unroll
            for (int r = 0; r < 8; r++) {
                const size_t row = (size_t)(rowBase + r + 8 * hi);
                __builtin_nontemporal_store((float)C4[r] * ko + bv, &out[row * 12 + ln]);
            }
        }
    }
}

extern "C" void kernel_launch(void* const* d_in, const int* in_sizes, int n_in,
                              void* d_out, int out_size, void* d_ws, size_t ws_size,
                              hipStream_t stream) {
    (void)in_sizes; (void)n_in; (void)out_size; (void)ws_size;
    const float* x  = (const float*)d_in[0];
    const float* W1 = (const float*)d_in[1];
    const float* g1 = (const float*)d_in[2];
    const float* b1 = (const float*)d_in[3];
    const float* s1 = (const float*)d_in[4];
    const float* W2 = (const float*)d_in[5];
    const float* g2 = (const float*)d_in[6];
    const float* b2 = (const float*)d_in[7];
    const float* s2 = (const float*)d_in[8];
    const float* W3 = (const float*)d_in[9];
    const float* g3 = (const float*)d_in[10];
    const float* b3 = (const float*)d_in[11];
    const float* s3 = (const float*)d_in[12];
    const float* Wo = (const float*)d_in[13];
    const float* bo = (const float*)d_in[14];
    uint8_t* ws = (uint8_t*)d_ws;

    qdense_prep<<<4, 256, 0, stream>>>(W1, W2, W3, Wo, ws);
    qdense_mlp<<<131072 / 128, 256, 0, stream>>>(x,
                                                 g1, b1, s1,
                                                 g2, b2, s2,
                                                 g3, b3, s3,
                                                 bo, ws, (float*)d_out);
}
